// Net_10342281249322
// MI455X (gfx1250) — matrix operand order repaired and checked by reading, not compiled, not run on hardware
//
#include <hip/hip_runtime.h>
#include <hip/hip_bf16.h>
#include <stdint.h>

#define TSEQ 512
#define NB   128
#define NI   512
#define NH   1024
#define G4H  4096
#define NCLS 1000

// step-kernel LDS staging: K processed in quarters
#define KQ      256
#define BROWS   64            // 4 gates x 16 columns of W_hh
#define BSTRIDE (KQ + 8)      // +8 bf16 pad -> 132-word row stride (bank stagger)

typedef __attribute__((ext_vector_type(16))) __bf16 v16bf;
typedef __attribute__((ext_vector_type(8)))  float  v8f;

union ABu { v16bf v; uint4 u[2]; };

__device__ __forceinline__ float sigf(float x) { return 1.0f / (1.0f + expf(-x)); }

__device__ __forceinline__ uint16_t f2bf_rne(float f) {
    union { float f; uint32_t u; } t; t.f = f;
    return (uint16_t)((t.u + 0x7FFFu + ((t.u >> 16) & 1u)) >> 16);
}

// ---------------- utility kernels ----------------

__global__ void f32_to_bf16_kernel(const float* __restrict__ s, uint16_t* __restrict__ d, int n) {
    int i = blockIdx.x * blockDim.x + threadIdx.x;
    if (i < n) d[i] = f2bf_rne(s[i]);
}

__global__ void zero_u32_kernel(uint32_t* __restrict__ p, int n) {
    int i = blockIdx.x * blockDim.x + threadIdx.x;
    if (i < n) p[i] = 0u;
}

// ---------------- phase 1: xproj = x @ W_ih.T + b_ih + b_hh ----------------
// Each wave computes a 16x64 output tile: one A (16x32) load feeds 4 WMMAs.
// grid (4096, 16), block 128 (4 waves); wave w -> ntiles [blockIdx.y*16 + w*4, +4)
__global__ __launch_bounds__(128) void xproj_gemm_kernel(
    const uint16_t* __restrict__ x_bf,    // [65536, 512] bf16
    const uint16_t* __restrict__ Wih_bf,  // [4096, 512]  bf16
    const float* __restrict__ b_ih,
    const float* __restrict__ b_hh,
    float* __restrict__ xproj)            // [65536, 4096] f32
{
    const int tid  = threadIdx.x;
    const int wave = tid >> 5;
    const int lane = tid & 31;
    const int ln15 = lane & 15;
    const int hi   = lane >> 4;

    const int mt     = blockIdx.x;                 // 0..4095
    const int ntbase = blockIdx.y * 16 + wave * 4; // 0..252 step 4

    const uint16_t* aptr = x_bf + (size_t)(mt * 16 + ln15) * NI + hi * 8;

    const uint16_t* bptr[4];
    int ncol[4];
    v8f acc[4];
#pragma unroll
    for (int j = 0; j < 4; ++j) {
        ncol[j] = (ntbase + j) * 16 + ln15;
        bptr[j] = Wih_bf + (size_t)ncol[j] * NI + hi * 8;
        float bias = b_ih[ncol[j]] + b_hh[ncol[j]];
#pragma unroll
        for (int v = 0; v < 8; ++v) acc[j][v] = bias;
    }

    for (int k = 0; k < NI; k += 32) {
        ABu a;
        a.u[0] = *(const uint4*)(aptr + k);
        a.u[1] = *(const uint4*)(aptr + k + 16);
#pragma unroll
        for (int j = 0; j < 4; ++j) {
            ABu b;
            b.u[0] = *(const uint4*)(bptr[j] + k);
            b.u[1] = *(const uint4*)(bptr[j] + k + 16);
            acc[j] = __builtin_amdgcn_wmma_f32_16x16x32_bf16(false, a.v, false, b.v,
                                                             (short)0, acc[j], false, false);
        }
    }

#pragma unroll
    for (int j = 0; j < 4; ++j) {
        float* dp = xproj + (size_t)(mt * 16 + hi * 8) * G4H + ncol[j];
#pragma unroll
        for (int v = 0; v < 8; ++v) dp[(size_t)v * G4H] = acc[j][v];
    }
}

// ---------------- phase 2: one LSTM timestep ----------------
// Block = one hidden-tile (ht), 8 waves = 8 batch-tiles. The block's 64 W_hh
// rows (4 gates x 16 cols) are staged into LDS once per K-quarter and shared
// by all 8 waves -> W_hh is read from L2 exactly once per step (8x less L2
// traffic than per-wave fetching). Each wave keeps all 4 gate accumulators,
// so the cell update stays pure register math (no gate exchange).
// grid (64), block 256 (8 waves).
__global__ __launch_bounds__(256) void lstm_step_kernel(
    const uint16_t* __restrict__ h_prev_bf,  // [128,1024] bf16
    const uint16_t* __restrict__ Whh_bf,     // [4096,1024] bf16
    const float* __restrict__ xproj_t,       // [128,4096] f32 (this step)
    const float* __restrict__ xproj_next,    // [128,4096] f32 (next step, prefetch)
    float* __restrict__ c,                   // [128,1024] f32
    float* __restrict__ h,                   // [128,1024] f32
    uint16_t* __restrict__ h_new_bf)         // [128,1024] bf16
{
    __shared__ alignas(16) uint16_t Bp[BROWS * BSTRIDE];  // 33 KB

    const int tid  = threadIdx.x;
    const int wave = tid >> 5;               // batch tile 0..7
    const int lane = tid & 31;
    const int ln15 = lane & 15;
    const int hi   = lane >> 4;

    const int bt = wave;
    const int ht = blockIdx.x;               // 0..63

    const uint16_t* aptr = h_prev_bf + (size_t)(bt * 16 + ln15) * NH + hi * 8;

    // seed accumulators from xproj (bias already folded in) + prefetch next step
    size_t cpoff[4];
    v8f acc[4];
#pragma unroll
    for (int g = 0; g < 4; ++g) {
        int ncol = g * NH + ht * 16 + ln15;
        cpoff[g] = (size_t)(bt * 16 + hi * 8) * G4H + ncol;
        const float* cp = xproj_t + cpoff[g];
#pragma unroll
        for (int v = 0; v < 8; ++v) acc[g][v] = cp[(size_t)v * G4H];
    }
#pragma unroll
    for (int g = 0; g < 4; ++g) {
        const float* pn = xproj_next + cpoff[g];
#pragma unroll
        for (int v = 0; v < 8; ++v) __builtin_prefetch(pn + (size_t)v * G4H, 0, 0);
    }

    for (int kq = 0; kq < NH; kq += KQ) {
        __syncthreads();   // previous quarter's readers are done
        // cooperative stage: 64 rows x 256 bf16 = 2048 uint4, 8 per thread
#pragma unroll
        for (int i = 0; i < 8; ++i) {
            int idx = tid + i * 256;          // 0..2047
            int rl  = idx >> 5;               // panel row 0..63
            int cch = idx & 31;               // 16B chunk 0..31
            int grow = (rl >> 4) * NH + ht * 16 + (rl & 15);   // global W_hh row
            uint4 d = *(const uint4*)(Whh_bf + (size_t)grow * NH + kq + cch * 8);
            *(uint4*)(Bp + rl * BSTRIDE + cch * 8) = d;
        }
        __syncthreads();

#pragma unroll 2
        for (int k = 0; k < KQ; k += 32) {
            ABu a;
            a.u[0] = *(const uint4*)(aptr + kq + k);
            a.u[1] = *(const uint4*)(aptr + kq + k + 16);
#pragma unroll
            for (int g = 0; g < 4; ++g) {
                const uint16_t* bp = Bp + (g * 16 + ln15) * BSTRIDE + hi * 8 + k;
                ABu b;
                b.u[0] = *(const uint4*)(bp);
                b.u[1] = *(const uint4*)(bp + 16);
                acc[g] = __builtin_amdgcn_wmma_f32_16x16x32_bf16(false, a.v, false, b.v,
                                                                 (short)0, acc[g], false, false);
            }
        }
    }

    // cell update entirely in registers: lane holds (m = hi*8+v, n = ln15)
#pragma unroll
    for (int v = 0; v < 8; ++v) {
        int off = (bt * 16 + hi * 8 + v) * NH + ht * 16 + ln15;
        float ig = sigf(acc[0][v]);
        float fg = sigf(acc[1][v]);
        float gg = tanhf(acc[2][v]);
        float og = sigf(acc[3][v]);
        float cn = fg * c[off] + ig * gg;
        c[off] = cn;
        float hn = og * tanhf(cn);
        h[off] = hn;
        h_new_bf[off] = f2bf_rne(hn);
    }
}

// ---------------- phase 3: out = h_last @ W_fc.T + b_fc (tiny, fp32) ----------------
__global__ void fc_kernel(const float* __restrict__ h, const float* __restrict__ Wfc,
                          const float* __restrict__ bfc, float* __restrict__ out)
{
    int idx = blockIdx.x * blockDim.x + threadIdx.x;
    if (idx >= NB * NCLS) return;
    int b = idx / NCLS, cls = idx - b * NCLS;
    const float* hp = h + (size_t)b * NH;
    const float* wp = Wfc + (size_t)cls * NH;
    float s = bfc[cls];
#pragma unroll 4
    for (int k = 0; k < NH; ++k) s += hp[k] * wp[k];
    out[idx] = s;
}

// ---------------- launch ----------------

extern "C" void kernel_launch(void* const* d_in, const int* in_sizes, int n_in,
                              void* d_out, int out_size, void* d_ws, size_t ws_size,
                              hipStream_t stream) {
    (void)in_sizes; (void)n_in; (void)out_size; (void)ws_size;

    const float* x    = (const float*)d_in[0];
    const float* W_ih = (const float*)d_in[1];
    const float* W_hh = (const float*)d_in[2];
    const float* b_ih = (const float*)d_in[3];
    const float* b_hh = (const float*)d_in[4];
    const float* W_fc = (const float*)d_in[5];
    const float* b_fc = (const float*)d_in[6];
    float* out = (float*)d_out;

    char* ws = (char*)d_ws;
    size_t off = 0;
    auto carve = [&](size_t bytes) { char* p = ws + off; off = (off + bytes + 255) & ~(size_t)255; return p; };

    uint16_t* x_bf    = (uint16_t*)carve((size_t)TSEQ * NB * NI * 2);     //  67 MB
    uint16_t* Wih_bf  = (uint16_t*)carve((size_t)G4H * NI * 2);           // 4.2 MB
    uint16_t* Whh_bf  = (uint16_t*)carve((size_t)G4H * NH * 2);           // 8.4 MB
    float*    xproj   = (float*)   carve((size_t)TSEQ * NB * G4H * 4);    // 1.07 GB
    float*    c_st    = (float*)   carve((size_t)NB * NH * 4);
    float*    h_st    = (float*)   carve((size_t)NB * NH * 4);
    uint16_t* hbf0    = (uint16_t*)carve((size_t)NB * NH * 2);
    uint16_t* hbf1    = (uint16_t*)carve((size_t)NB * NH * 2);

    // ---- convert fp32 -> bf16 ----
    {
        int n = TSEQ * NB * NI;
        f32_to_bf16_kernel<<<(n + 255) / 256, 256, 0, stream>>>(x, x_bf, n);
        n = G4H * NI;
        f32_to_bf16_kernel<<<(n + 255) / 256, 256, 0, stream>>>(W_ih, Wih_bf, n);
        n = G4H * NH;
        f32_to_bf16_kernel<<<(n + 255) / 256, 256, 0, stream>>>(W_hh, Whh_bf, n);
    }

    // ---- zero initial state: c = 0 (f32), h0_bf16 = 0 ----
    {
        int n = NB * NH;
        zero_u32_kernel<<<(n + 255) / 256, 256, 0, stream>>>((uint32_t*)c_st, n);
        int n2 = NB * NH / 2;
        zero_u32_kernel<<<(n2 + 255) / 256, 256, 0, stream>>>((uint32_t*)hbf0, n2);
    }

    // ---- phase 1: big input GEMM ----
    {
        dim3 grid(TSEQ * NB / 16, G4H / (16 * 16));  // (4096, 16)
        xproj_gemm_kernel<<<grid, 128, 0, stream>>>(x_bf, Wih_bf, b_ih, b_hh, xproj);
    }

    // ---- phase 2: sequential recurrence, one kernel per timestep ----
    {
        for (int t = 0; t < TSEQ; ++t) {
            const uint16_t* hin  = (t & 1) ? hbf1 : hbf0;
            uint16_t*       hout = (t & 1) ? hbf0 : hbf1;
            const float* xp_t = xproj + (size_t)t * NB * G4H;
            const float* xp_n = (t + 1 < TSEQ) ? xproj + (size_t)(t + 1) * NB * G4H : xp_t;
            lstm_step_kernel<<<NH / 16, 256, 0, stream>>>(hin, Whh_bf, xp_t, xp_n,
                                                          c_st, h_st, hout);
        }
    }

    // ---- phase 3: final FC ----
    {
        int n = NB * NCLS;
        fc_kernel<<<(n + 255) / 256, 256, 0, stream>>>(h_st, W_fc, b_fc, out);
    }
}
